// XConvAutoEncoder_25529285607673
// MI455X (gfx1250) — matrix operand order repaired and checked by reading, not compiled, not run on hardware
//
#include <hip/hip_runtime.h>
#include <math.h>
#include <stdint.h>

#define NPTS 8192
#define KNN  10

typedef __attribute__((ext_vector_type(16))) __bf16 v16bf;
typedef __attribute__((ext_vector_type(8)))  float  v8f;

#if __has_builtin(__builtin_amdgcn_global_load_async_to_lds_b128) && \
    __has_builtin(__builtin_amdgcn_s_wait_asynccnt)
#define USE_ASYNC_LDS 1
typedef int v4i_vec __attribute__((vector_size(16)));
typedef __attribute__((address_space(1))) v4i_vec gm_v4i;
typedef __attribute__((address_space(3))) v4i_vec lds_v4i;
#else
#define USE_ASYNC_LDS 0
#endif

__device__ __forceinline__ float elu1(float x) {
    return x > 0.f ? x : (__expf(x) - 1.f);
}

// Stage one 16-byte chunk of the A tile into LDS (async engine if available).
__device__ __forceinline__ void stage16(const __bf16* gsrc, unsigned int* ldst) {
#if USE_ASYNC_LDS
    __builtin_amdgcn_global_load_async_to_lds_b128(
        (gm_v4i*)(uintptr_t)gsrc,
        (lds_v4i*)(uint32_t)(uintptr_t)ldst,
        0, 0);
#else
    *(uint4*)ldst = *(const uint4*)gsrc;
#endif
}
__device__ __forceinline__ void stage_wait_pending() {   // allow 1 copy in flight
#if USE_ASYNC_LDS
    __builtin_amdgcn_s_wait_asynccnt(1);
#endif
}
__device__ __forceinline__ void stage_wait_all() {
#if USE_ASYNC_LDS
    __builtin_amdgcn_s_wait_asynccnt(0);
#endif
}

// ---------------------------------------------------------------------------
// KNN: one thread per query point, candidates tiled through LDS, top-10
// insertion sort in registers (incl. self, single batch).
// ---------------------------------------------------------------------------
__global__ void knn_kernel(const float* __restrict__ pos, int* __restrict__ idx, int N) {
    __shared__ float sp[256 * 3];
    int i = blockIdx.x * 256 + threadIdx.x;
    float px = pos[i * 3 + 0], py = pos[i * 3 + 1], pz = pos[i * 3 + 2];
    float bd[KNN];
    int   bi[KNN];
#pragma unroll
    for (int k = 0; k < KNN; ++k) { bd[k] = 3.4e38f; bi[k] = 0; }
    for (int j0 = 0; j0 < N; j0 += 256) {
        int j = j0 + threadIdx.x;
        sp[threadIdx.x * 3 + 0] = pos[j * 3 + 0];
        sp[threadIdx.x * 3 + 1] = pos[j * 3 + 1];
        sp[threadIdx.x * 3 + 2] = pos[j * 3 + 2];
        __syncthreads();
        for (int jj = 0; jj < 256; ++jj) {
            float dx = sp[jj * 3 + 0] - px;
            float dy = sp[jj * 3 + 1] - py;
            float dz = sp[jj * 3 + 2] - pz;
            float d = dx * dx + dy * dy + dz * dz;
            if (d < bd[KNN - 1]) {
                int p = KNN - 1;
                while (p > 0 && bd[p - 1] > d) {
                    bd[p] = bd[p - 1]; bi[p] = bi[p - 1]; --p;
                }
                bd[p] = d; bi[p] = j0 + jj;
            }
        }
        __syncthreads();
    }
#pragma unroll
    for (int k = 0; k < KNN; ++k) idx[i * KNN + k] = bi[k];
}

// ---------------------------------------------------------------------------
// Column batch statistics (BatchNorm training mode, biased variance).
// ---------------------------------------------------------------------------
__global__ void colstats_kernel(const float* __restrict__ x, int rows, int cols,
                                float* __restrict__ mean, float* __restrict__ rstd) {
    int c = blockIdx.x;
    int tid = threadIdx.x;
    float s = 0.f, s2 = 0.f;
    for (int r = tid; r < rows; r += 256) {
        float v = x[(size_t)r * cols + c];
        s += v; s2 += v * v;
    }
    __shared__ float sh[256], sh2[256];
    sh[tid] = s; sh2[tid] = s2;
    __syncthreads();
    for (int o = 128; o > 0; o >>= 1) {
        if (tid < o) { sh[tid] += sh[tid + o]; sh2[tid] += sh2[tid + o]; }
        __syncthreads();
    }
    if (tid == 0) {
        float m = sh[0] / (float)rows;
        float v = sh2[0] / (float)rows - m * m;
        mean[c] = m;
        rstd[c] = rsqrtf(v + 1e-5f);
    }
}

// ---------------------------------------------------------------------------
// mlp1 stage 1: rel positions + first Linear+ELU (pre-BN activations stored).
// ---------------------------------------------------------------------------
__global__ void mlp1_stage1_kernel(const float* __restrict__ pos, const int* __restrict__ idx,
                                   const float* __restrict__ w1, const float* __restrict__ b1,
                                   float* __restrict__ rel, float* __restrict__ pre1, int NK) {
    int t = blockIdx.x * blockDim.x + threadIdx.x;
    if (t >= NK) return;
    int n = t / KNN;
    int j = idx[t];
    float rx = pos[j * 3 + 0] - pos[n * 3 + 0];
    float ry = pos[j * 3 + 1] - pos[n * 3 + 1];
    float rz = pos[j * 3 + 2] - pos[n * 3 + 2];
    rel[t * 3 + 0] = rx; rel[t * 3 + 1] = ry; rel[t * 3 + 2] = rz;
#pragma unroll
    for (int c = 0; c < 3; ++c) {
        float z = b1[c] + w1[c * 3 + 0] * rx + w1[c * 3 + 1] * ry + w1[c * 3 + 2] * rz;
        pre1[t * 3 + c] = elu1(z);
    }
}

// mlp1 stage 2: BN1 apply + second Linear+ELU (pre-BN2 stored).
__global__ void mlp1_stage2_kernel(const float* __restrict__ pre1,
                                   const float* __restrict__ m1, const float* __restrict__ r1,
                                   const float* __restrict__ g1, const float* __restrict__ be1,
                                   const float* __restrict__ w2, const float* __restrict__ b2,
                                   float* __restrict__ pre2, int NK) {
    int t = blockIdx.x * blockDim.x + threadIdx.x;
    if (t >= NK) return;
    float h[3];
#pragma unroll
    for (int c = 0; c < 3; ++c)
        h[c] = (pre1[t * 3 + c] - m1[c]) * r1[c] * g1[c] + be1[c];
#pragma unroll
    for (int c = 0; c < 3; ++c) {
        float z = b2[c] + w2[c * 3 + 0] * h[0] + w2[c * 3 + 1] * h[1] + w2[c * 3 + 2] * h[2];
        pre2[t * 3 + c] = elu1(z);
    }
}

// X-transform stage 1: Linear(K*D -> K*K) + ELU.
__global__ void t_stage1_kernel(const float* __restrict__ rel,
                                const float* __restrict__ wt, const float* __restrict__ bt,
                                float* __restrict__ pre_t, int total) {
    int t = blockIdx.x * blockDim.x + threadIdx.x;
    if (t >= total) return;
    int n = t / 100, j = t % 100;
    const float* r = rel + (size_t)n * 30;
    const float* w = wt + (size_t)j * 30;
    float acc = bt[j];
#pragma unroll
    for (int l = 0; l < 30; ++l) acc += w[l] * r[l];
    pre_t[t] = elu1(acc);
}

// X-transform grouped stage: BN(in) -> per-group KxK linear (+bias) -> [ELU].
__global__ void t_qstage_kernel(const float* __restrict__ tin,
                                const float* __restrict__ mean, const float* __restrict__ rstd,
                                const float* __restrict__ gamma, const float* __restrict__ beta,
                                const float* __restrict__ wq, const float* __restrict__ bq,
                                float* __restrict__ tout, int total, int do_elu) {
    int t = blockIdx.x * blockDim.x + threadIdx.x;
    if (t >= total) return;
    int n = t / 100, j = t % 100;
    int g = j / 10, jj = j % 10;
    const float* base = tin + (size_t)n * 100 + g * 10;
    const float* w = wq + g * 100 + jj * 10;
    int cb = g * 10;
    float acc = bq[j];
#pragma unroll
    for (int l = 0; l < 10; ++l) {
        float v = (base[l] - mean[cb + l]) * rstd[cb + l] * gamma[cb + l] + beta[cb + l];
        acc += w[l] * v;
    }
    tout[t] = do_elu ? elu1(acc) : acc;
}

// ---------------------------------------------------------------------------
// XConv core: BN-final on X-transform, build x_star on the fly, xt = x_star@t,
// depthwise conv. Emits the feature matrix DIRECTLY as zero-padded bf16
// [N, Kpad] so the WMMA GEMM runs guard-free.
// ---------------------------------------------------------------------------
__global__ void xconv_final_kernel(const float* __restrict__ pq2,
                                   const float* __restrict__ mq, const float* __restrict__ rq,
                                   const float* __restrict__ gq, const float* __restrict__ bq,
                                   const float* __restrict__ pre2,
                                   const float* __restrict__ m2, const float* __restrict__ r2,
                                   const float* __restrict__ g2, const float* __restrict__ be2,
                                   const float* __restrict__ xprev, const int* __restrict__ idx,
                                   const float* __restrict__ wc, const float* __restrict__ bc,
                                   __bf16* __restrict__ featb, int CIN, int DM, int Kpad) {
    int n = blockIdx.x;
    int tid = threadIdx.x;
    __shared__ float T[100];
    __shared__ float XS[30];          // [c*10+k] mlp1-part of x_star
    if (tid < 100)
        T[tid] = (pq2[(size_t)n * 100 + tid] - mq[tid]) * rq[tid] * gq[tid] + bq[tid];
    if (tid >= 100 && tid < 130) {
        int u = tid - 100, k = u / 3, c = u % 3;
        XS[c * 10 + k] = (pre2[((size_t)n * KNN + k) * 3 + c] - m2[c]) * r2[c] * g2[c] + be2[c];
    }
    __syncthreads();
    int CC = CIN + 3;
    int F = CC * DM;
    __bf16* frow = featb + (size_t)n * Kpad;
    for (int c = tid; c < CC; c += blockDim.x) {
        float row[KNN];
        if (c < 3) {
#pragma unroll
            for (int k = 0; k < KNN; ++k) row[k] = XS[c * 10 + k];
        } else {
#pragma unroll
            for (int k = 0; k < KNN; ++k)
                row[k] = xprev[(size_t)idx[n * KNN + k] * CIN + (c - 3)];
        }
        float xt[KNN];
#pragma unroll
        for (int j = 0; j < KNN; ++j) {
            float a = 0.f;
#pragma unroll
            for (int k = 0; k < KNN; ++k) a += row[k] * T[k * 10 + j];
            xt[j] = a;
        }
        for (int m = 0; m < DM; ++m) {
            const float* w = wc + ((size_t)c * DM + m) * KNN;
            float a = bc[c * DM + m];
#pragma unroll
            for (int k = 0; k < KNN; ++k) a += xt[k] * w[k];
            frow[c * DM + m] = (__bf16)a;
        }
    }
    // zero-fill K padding so GEMM needs no guards
    for (int p = F + tid; p < Kpad; p += blockDim.x) frow[p] = (__bf16)0.f;
}

// ---------------------------------------------------------------------------
// Pack wl[COUT, Kd] (f32) -> k-major bf16 [Kpad, COUT], zero pad rows.
// ---------------------------------------------------------------------------
__global__ void packW_kernel(const float* __restrict__ wl, __bf16* __restrict__ wt,
                             int Kd, int Kpad, int COUT) {
    int t = blockIdx.x * blockDim.x + threadIdx.x;
    if (t >= Kpad * COUT) return;
    int k = t / COUT, o = t % COUT;
    wt[t] = (k < Kd) ? (__bf16)wl[(size_t)o * Kd + k] : (__bf16)0.f;
}

// ---------------------------------------------------------------------------
// WMMA GEMM: Y[M,COUT] = Ab[M,Kpad] @ Bt[Kpad,COUT] + bias (optional ReLU).
// Ab row-major bf16 (zero-padded), Bt k-major bf16 (already W^T).
// Block = 256 thr (8 waves); block tile 32 rows x 64 cols; wave = 16x16 tile.
// A tile (32x64 bf16) double-buffered in LDS, staged by the async global->LDS
// engine when available (ASYNCcnt) so the next copy overlaps current WMMAs;
// B fragments are direct per-element bf16 global loads. Guard-free inner loop, two
// v_wmma_f32_16x16x32_bf16 per K-step.
// ---------------------------------------------------------------------------
__global__ void gemm_wmma_kernel(const __bf16* __restrict__ Ab, const __bf16* __restrict__ Bt,
                                 const float* __restrict__ bias, float* __restrict__ Y,
                                 int Kpad, int COUT, int relu) {
    __shared__ unsigned int As[2][32 * 36];   // 32 rows x 32 u32 (64 bf16), stride 36
    int tid  = threadIdx.x;
    int lane = tid & 31;
    int wv   = tid >> 5;
    int row0 = blockIdx.x * 32 + (wv >> 2) * 16;
    int col0 = blockIdx.y * 64 + (wv & 3) * 16;
    int m = lane & 15;
    int g = lane >> 4;

    int cr = tid >> 3;                        // coop-stage row 0..31
    int cq = tid & 7;                         // 16B chunk within 128B row
    int arow0 = blockIdx.x * 32;
    const __bf16* gsrc_base = Ab + (size_t)(arow0 + cr) * Kpad + cq * 8;
    unsigned int* lds_slot0 = &As[0][cr * 36 + cq * 4];
    unsigned int* lds_slot1 = &As[1][cr * 36 + cq * 4];

    // A-fragment u32 pair offsets within a 64-wide k-row (per ISA layout):
    // v<4 -> v + 4g ; v>=4 -> 8 + (v-4) + 4g ; +16 for the second k-half.
    int pbase[8];
#pragma unroll
    for (int v = 0; v < 8; ++v)
        pbase[v] = (v < 4) ? (v + 4 * g) : (8 + (v - 4) + 4 * g);

    // prologue: stage tile 0 into buffer 0
    stage16(gsrc_base, lds_slot0);

    v8f acc = {};
    int it = 0;
    for (int kk = 0; kk < Kpad; kk += 64, ++it) {
        unsigned int* curbuf = (it & 1) ? &As[1][0] : &As[0][0];
        if (kk + 64 < Kpad) {
            stage16(gsrc_base + kk + 64, (it & 1) ? lds_slot0 : lds_slot1);
            stage_wait_pending();
        } else {
            stage_wait_all();
        }
        __syncthreads();

        union { v16bf v; unsigned int u[8]; } a0, a1;
        union { v16bf v; uint4 q[2];       } b0, b1;
        const unsigned int* arow = curbuf + (((wv >> 2) * 16) + m) * 36;
#pragma unroll
        for (int v = 0; v < 8; ++v) {
            a0.u[v] = arow[pbase[v]];
            a1.u[v] = arow[pbase[v] + 16];
        }
#pragma unroll
        for (int s = 0; s < 16; ++s) {
            int kb = (s < 8) ? (8 * g + s) : (16 + 8 * g + (s - 8));
            b0.v[s] = Bt[(size_t)(kk + kb) * COUT + col0 + m];
            b1.v[s] = Bt[(size_t)(kk + 32 + kb) * COUT + col0 + m];
        }

        acc = __builtin_amdgcn_wmma_f32_16x16x32_bf16(
            false, a0.v, false, b0.v, (short)0, acc, false, false);
        acc = __builtin_amdgcn_wmma_f32_16x16x32_bf16(
            false, a1.v, false, b1.v, (short)0, acc, false, false);
        __syncthreads();                      // buffer reuse fence
    }

    // C/D layout: VGPR r, lanes0-15 -> M=r, lanes16-31 -> M=r+8; N = lane&15
#pragma unroll
    for (int r = 0; r < 8; ++r) {
        int row = row0 + r + 8 * g;
        int col = col0 + m;
        float y = acc[r] + bias[col];
        if (relu) y = fmaxf(y, 0.f);
        Y[(size_t)row * COUT + col] = y;
    }
}

// ---------------------------------------------------------------------------
// Global max-pool over points: one block per channel.
// ---------------------------------------------------------------------------
__global__ void maxreduce_kernel(const float* __restrict__ x, float* __restrict__ enc,
                                 int N, int C) {
    int c = blockIdx.x, tid = threadIdx.x;
    float mv = -3.4e38f;
    for (int r = tid; r < N; r += 256) mv = fmaxf(mv, x[(size_t)r * C + c]);
    __shared__ float sh[256];
    sh[tid] = mv;
    __syncthreads();
    for (int o = 128; o > 0; o >>= 1) {
        if (tid < o) sh[tid] = fmaxf(sh[tid], sh[tid + o]);
        __syncthreads();
    }
    if (tid == 0) enc[c] = sh[0];
}

// Decoder GEMVs: d1 = relu(dw1 @ enc); d2 = dw2 @ d1. Single block, 256 thr.
__global__ void decoder_kernel(const float* __restrict__ enc,
                               const float* __restrict__ dw1, const float* __restrict__ dw2,
                               float* __restrict__ d2) {
    __shared__ float e[128];
    __shared__ float d1[256];
    int tid = threadIdx.x;
    if (tid < 128) e[tid] = enc[tid];
    __syncthreads();
    float a = 0.f;
    for (int j = 0; j < 128; ++j) a += dw1[(size_t)tid * 128 + j] * e[j];
    d1[tid] = fmaxf(a, 0.f);
    __syncthreads();
    float b = 0.f;
    for (int j = 0; j < 256; ++j) b += dw2[(size_t)tid * 256 + j] * d1[j];
    d2[tid] = b;
}

// Final: d = dw3 @ d2 + db3, reshape(3,N).T -> out[n*3+c].
__global__ void decoder_out_kernel(const float* __restrict__ dw3, const float* __restrict__ db3,
                                   const float* __restrict__ d2, float* __restrict__ out, int N) {
    __shared__ float d[256];
    int tid = threadIdx.x;
    d[tid] = d2[tid];
    __syncthreads();
    int r = blockIdx.x * 256 + tid;
    const float* w = dw3 + (size_t)r * 256;
    float a = db3[r];
    for (int j = 0; j < 256; ++j) a += w[j] * d[j];
    int c = r / N, n = r % N;
    out[n * 3 + c] = a;
}

// ---------------------------------------------------------------------------
extern "C" void kernel_launch(void* const* d_in, const int* in_sizes, int n_in,
                              void* d_out, int out_size, void* d_ws, size_t ws_size,
                              hipStream_t stream) {
    (void)in_sizes; (void)n_in; (void)out_size; (void)ws_size;
    const int N = NPTS;
    const int NK = N * KNN;
    const float* pos = (const float*)d_in[0];
    // d_in[1] = batch (all zeros, single batch) — unused.
    const int CIN[5]  = {3, 64, 128, 128, 256};
    const int COUT[5] = {64, 128, 128, 256, 128};

    // ---- workspace carve-up ----
    char* wsb = (char*)d_ws;
    size_t off = 0;
    auto take = [&](size_t bytes) -> void* {
        void* p = wsb + off;
        off += (bytes + 255) & ~(size_t)255;
        return p;
    };
    int*    idx   = (int*)   take((size_t)NK * sizeof(int));
    float*  rel   = (float*) take((size_t)NK * 3 * sizeof(float));
    float*  pre1  = (float*) take((size_t)NK * 3 * sizeof(float));
    float*  pre2  = (float*) take((size_t)NK * 3 * sizeof(float));
    float*  pt    = (float*) take((size_t)N * 100 * sizeof(float));
    float*  eq1   = (float*) take((size_t)N * 100 * sizeof(float));
    float*  pq2   = (float*) take((size_t)N * 100 * sizeof(float));
    float*  xa    = (float*) take((size_t)N * 256 * sizeof(float));
    float*  xb    = (float*) take((size_t)N * 256 * sizeof(float));
    __bf16* featb = (__bf16*)take((size_t)N * 320 * sizeof(__bf16));   // max Kpad
    __bf16* wltb  = (__bf16*)take((size_t)320 * 256 * sizeof(__bf16)); // max Kpad*COUT
    float*  st    = (float*) take(1024 * sizeof(float));
    float* m1 = st + 0,   *r1 = st + 3,   *m2 = st + 6,   *r2 = st + 9;
    float* mt = st + 12,  *rt = st + 112;
    float* mq1 = st + 212, *rq1 = st + 312;
    float* mq2 = st + 412, *rq2 = st + 512;
    float* enc = st + 612;
    float* d2v = st + 744;

    knn_kernel<<<N / 256, 256, 0, stream>>>(pos, idx, N);

    const float* xprev = pos;
    float* bufs[2] = {xa, xb};
    for (int l = 0; l < 5; ++l) {
        int base = 2 + 24 * l;
        const float* w1  = (const float*)d_in[base + 0];
        const float* b1  = (const float*)d_in[base + 1];
        const float* g1  = (const float*)d_in[base + 2];
        const float* be1 = (const float*)d_in[base + 3];
        const float* w2  = (const float*)d_in[base + 4];
        const float* b2  = (const float*)d_in[base + 5];
        const float* g2  = (const float*)d_in[base + 6];
        const float* be2 = (const float*)d_in[base + 7];
        const float* wt  = (const float*)d_in[base + 8];
        const float* bt  = (const float*)d_in[base + 9];
        const float* gt  = (const float*)d_in[base + 10];
        const float* bet = (const float*)d_in[base + 11];
        const float* wq1 = (const float*)d_in[base + 12];
        const float* bq1 = (const float*)d_in[base + 13];
        const float* gq1 = (const float*)d_in[base + 14];
        const float* beq1= (const float*)d_in[base + 15];
        const float* wq2 = (const float*)d_in[base + 16];
        const float* bq2 = (const float*)d_in[base + 17];
        const float* gq2 = (const float*)d_in[base + 18];
        const float* beq2= (const float*)d_in[base + 19];
        const float* wc  = (const float*)d_in[base + 20];
        const float* bc  = (const float*)d_in[base + 21];
        const float* wl  = (const float*)d_in[base + 22];
        const float* bl  = (const float*)d_in[base + 23];

        int cin = CIN[l], cout = COUT[l];
        int cc = cin + 3;
        int dm = (cout + cc - 1) / cc;
        int fdim = cc * dm;
        int kpad = ((fdim + 63) / 64) * 64;
        float* xout = bufs[l & 1];

        mlp1_stage1_kernel<<<(NK + 255) / 256, 256, 0, stream>>>(pos, idx, w1, b1, rel, pre1, NK);
        colstats_kernel<<<3, 256, 0, stream>>>(pre1, NK, 3, m1, r1);
        mlp1_stage2_kernel<<<(NK + 255) / 256, 256, 0, stream>>>(pre1, m1, r1, g1, be1, w2, b2, pre2, NK);
        colstats_kernel<<<3, 256, 0, stream>>>(pre2, NK, 3, m2, r2);

        t_stage1_kernel<<<(N * 100 + 255) / 256, 256, 0, stream>>>(rel, wt, bt, pt, N * 100);
        colstats_kernel<<<100, 256, 0, stream>>>(pt, N, 100, mt, rt);
        t_qstage_kernel<<<(N * 100 + 255) / 256, 256, 0, stream>>>(pt, mt, rt, gt, bet, wq1, bq1, eq1, N * 100, 1);
        colstats_kernel<<<100, 256, 0, stream>>>(eq1, N, 100, mq1, rq1);
        t_qstage_kernel<<<(N * 100 + 255) / 256, 256, 0, stream>>>(eq1, mq1, rq1, gq1, beq1, wq2, bq2, pq2, N * 100, 0);
        colstats_kernel<<<100, 256, 0, stream>>>(pq2, N, 100, mq2, rq2);

        xconv_final_kernel<<<N, 256, 0, stream>>>(pq2, mq2, rq2, gq2, beq2,
                                                  pre2, m2, r2, g2, be2,
                                                  xprev, idx, wc, bc, featb, cin, dm, kpad);

        packW_kernel<<<(kpad * cout + 255) / 256, 256, 0, stream>>>(wl, wltb, fdim, kpad, cout);

        dim3 gg(N / 32, cout / 64);
        gemm_wmma_kernel<<<gg, 256, 0, stream>>>(featb, wltb, bl, xout, kpad, cout, (l < 4) ? 1 : 0);
        xprev = xout;
    }

    const float* dw1 = (const float*)d_in[122];
    const float* dw2 = (const float*)d_in[123];
    const float* dw3 = (const float*)d_in[124];
    const float* db3 = (const float*)d_in[125];

    maxreduce_kernel<<<128, 256, 0, stream>>>(xprev, enc, N, 128);
    decoder_kernel<<<1, 256, 0, stream>>>(enc, dw1, dw2, d2v);
    decoder_out_kernel<<<(3 * N) / 256, 256, 0, stream>>>(dw3, db3, d2v, (float*)d_out, N);
}
